// Petra_85066122264991
// MI455X (gfx1250) — matrix operand order repaired and checked by reading, not compiled, not run on hardware
//
#include <hip/hip_runtime.h>
#include <hip/hip_bf16.h>
#include <math.h>

// ---------------------------------------------------------------------------
// Problem constants (from the reference)
// ---------------------------------------------------------------------------
constexpr int B_    = 2;
constexpr int S_    = 1024;     // src length
constexpr int L_    = 767;      // tgt half length (before cls)
constexpr int T_    = 2 * (L_ + 1);   // 1536 concatenated target tokens
constexpr int D_    = 256;      // d_model
constexpr int H_    = 8;        // heads
constexpr int DH_   = 2048;     // dim per head
constexpr int INNER_= H_ * DH_; // 16384
constexpr int V_    = 25426;    // vocab
constexpr int CLS_  = V_;       // cls token id
constexpr int MASK_ = V_ + 2;   // mask token id
constexpr int BT_   = B_ * T_;  // 3072 rows

constexpr long BTD_  = (long)BT_ * D_;        // 786432
constexpr long TI_   = (long)BT_ * INNER_;    // 50331648

// ---------------------------------------------------------------------------
// CDNA5 WMMA / packing types
// ---------------------------------------------------------------------------
typedef __attribute__((ext_vector_type(16))) __bf16        v16bf;
typedef __attribute__((ext_vector_type(8)))  float         v8f;
typedef unsigned int                                        u32;
typedef __attribute__((ext_vector_type(2)))  unsigned int  u32x2;
typedef __attribute__((ext_vector_type(8)))  unsigned int  u32x8;
typedef __attribute__((ext_vector_type(4)))  float         f32x4;

#if defined(__has_builtin)
#if __has_builtin(__builtin_amdgcn_cvt_pk_bf16_f32)
#define HAVE_CVT_PK_BF16 1
#endif
#endif

// pack two fp32 into one u32 of two bf16
__device__ inline u32 pack_bf2(float lo, float hi) {
#ifdef HAVE_CVT_PK_BF16
    auto r = __builtin_amdgcn_cvt_pk_bf16_f32(lo, hi);
    return __builtin_bit_cast(u32, r);
#else
    u32 a = __builtin_bit_cast(u32, lo);
    u32 b = __builtin_bit_cast(u32, hi);
    a += 0x7fffu + ((a >> 16) & 1u);        // round-to-nearest-even
    b += 0x7fffu + ((b >> 16) & 1u);
    return (a >> 16) | (b & 0xffff0000u);
#endif
}

// ---------------------------------------------------------------------------
// Reductions (wave32)
// ---------------------------------------------------------------------------
__device__ inline float wave_sum(float v) {
#pragma unroll
    for (int o = 16; o > 0; o >>= 1) v += __shfl_xor(v, o, 32);
    return v;
}
__device__ inline float wave_max(float v) {
#pragma unroll
    for (int o = 16; o > 0; o >>= 1) v = fmaxf(v, __shfl_xor(v, o, 32));
    return v;
}
__device__ inline float block_sum(float v, float* sh) { // blockDim.x == 256
    int tid = threadIdx.x, wv = tid >> 5, ln = tid & 31;
    v = wave_sum(v);
    if (ln == 0) sh[wv] = v;
    __syncthreads();
    if (tid == 0) { float s = sh[0]; for (int i = 1; i < 8; ++i) s += sh[i]; sh[0] = s; }
    __syncthreads();
    float r = sh[0];
    __syncthreads();
    return r;
}
__device__ inline float block_max(float v, float* sh) {
    int tid = threadIdx.x, wv = tid >> 5, ln = tid & 31;
    v = wave_max(v);
    if (ln == 0) sh[wv] = v;
    __syncthreads();
    if (tid == 0) { float s = sh[0]; for (int i = 1; i < 8; ++i) s = fmaxf(s, sh[i]); sh[0] = s; }
    __syncthreads();
    float r = sh[0];
    __syncthreads();
    return r;
}

// ---------------------------------------------------------------------------
// Generic batched GEMM on the bf16 WMMA pipe, fp32 accumulate.
//   C[z][m][n] = act( alpha * sum_k A[z][m][k] * B'[z][k][n] + bias[n] )
//   B' = B ([K,N] row-major) when !TRANS_B, else B is [N,K] (C = A*B^T).
//   Batch z = zo*Hdiv + zh with independent strides (serves per-(b,h) GEMMs).
//
// Block tile 128(M) x 64(N) x 32(K); 8 waves in a 4x2 grid, each wave owns a
// 2x2 arrangement of 16x16 WMMA tiles -> A/B fragments each reused twice,
// 4 v_wmma per wave per K stage.
//
// Call-site guarantees: M % 128 == 0, K % 32 == 0; N % 64 == 0 for TRANS_B.
// Ragged N (logits vocab) handled by block-uniform fullN path.
// LDS: bf16 pairs packed in u32, B stored K-packed per N row so fragments are
// contiguous ds_load_b64 runs; row stride 18 u32 spreads banks.
// ---------------------------------------------------------------------------
template<bool TRANS_B, int ACT>   // ACT: 0 = none, 1 = exact GELU
__global__ __launch_bounds__(256)
void gemm_wmma_kernel(const float* __restrict__ Ag, const float* __restrict__ Bg,
                      const float* __restrict__ bias, float* __restrict__ Cg,
                      int M, int N, int K, int lda, int ldb, int ldc,
                      int Hdiv, long sAo, long sAh, long sBo, long sBh,
                      long sCo, long sCh, float alpha)
{
    __shared__ u32 As[128][18];   // 128 M rows x 32 K (16 packed u32) + pad
    __shared__ u32 Bs[64][18];    // 64 N rows  x 32 K (16 packed u32) + pad

    const int z  = blockIdx.z;
    const int zo = z / Hdiv, zh = z % Hdiv;
    const float* A  = Ag + zo * sAo + (long)zh * sAh;
    const float* Bp = Bg + zo * sBo + (long)zh * sBh;
    float*       C  = Cg + zo * sCo + (long)zh * sCh;

    const int mbase = blockIdx.y * 128;
    const int nbase = blockIdx.x * 64;
    const bool fullN = (nbase + 64 <= N);        // block-uniform

    const int tid  = threadIdx.x;
    const int wave = tid >> 5;
    const int lane = tid & 31;
    const int hl   = lane >> 4;
    const int l    = lane & 15;
    const int wm   = wave & 3;    // M group of 32
    const int wn   = wave >> 2;   // N group of 32

    // staging thread mappings
    const int ar   = tid >> 1;             // A row 0..127
    const int ac   = (tid & 1) * 16;       // A K col (floats): 0 or 16
    const int bn_f = tid & 63;             // !TRANS_B: N col
    const int bj   = (tid >> 6) * 4;       // !TRANS_B: u32 group (K = 2*bj..2*bj+7)
    const int bn_t = tid >> 2;             // TRANS_B: N row 0..63
    const int bc_t = (tid & 3) * 8;        // TRANS_B: K col (floats)

    v8f acc[2][2] = {};

    for (int k0 = 0; k0 < K; k0 += 32) {
        // ---- stage A 128x32: 4x b128 loads, pack, 4x b64 LDS stores ----
        {
            const f32x4* ap = (const f32x4*)(A + (long)(mbase + ar) * lda + k0 + ac);
            f32x4 f0 = ap[0], f1 = ap[1], f2 = ap[2], f3 = ap[3];
            const int c0 = ac >> 1;
            u32x2 p;
            p.x = pack_bf2(f0.x, f0.y); p.y = pack_bf2(f0.z, f0.w);
            *(u32x2*)&As[ar][c0 + 0] = p;
            p.x = pack_bf2(f1.x, f1.y); p.y = pack_bf2(f1.z, f1.w);
            *(u32x2*)&As[ar][c0 + 2] = p;
            p.x = pack_bf2(f2.x, f2.y); p.y = pack_bf2(f2.z, f2.w);
            *(u32x2*)&As[ar][c0 + 4] = p;
            p.x = pack_bf2(f3.x, f3.y); p.y = pack_bf2(f3.z, f3.w);
            *(u32x2*)&As[ar][c0 + 6] = p;
        }
        // ---- stage B 32(K) x 64(N), K-packed per N row ----
        if (!TRANS_B) {
            const long gk = k0 + 2 * bj;
            if (fullN) {
                const float* bp = Bp + gk * ldb + nbase + bn_f;
                float v0 = bp[0],            v1 = bp[(long)ldb],
                      v2 = bp[2*(long)ldb],  v3 = bp[3*(long)ldb],
                      v4 = bp[4*(long)ldb],  v5 = bp[5*(long)ldb],
                      v6 = bp[6*(long)ldb],  v7 = bp[7*(long)ldb];
                u32x2 p;
                p.x = pack_bf2(v0, v1); p.y = pack_bf2(v2, v3);
                *(u32x2*)&Bs[bn_f][bj + 0] = p;
                p.x = pack_bf2(v4, v5); p.y = pack_bf2(v6, v7);
                *(u32x2*)&Bs[bn_f][bj + 2] = p;
            } else {                       // logits right edge only
                const int gc = nbase + bn_f;
                const bool in = gc < N;
                const float* bp = Bp + gk * ldb + (in ? gc : 0);
                float v[8];
#pragma unroll
                for (int i = 0; i < 8; ++i) v[i] = in ? bp[i * (long)ldb] : 0.0f;
                u32x2 p;
                p.x = pack_bf2(v[0], v[1]); p.y = pack_bf2(v[2], v[3]);
                *(u32x2*)&Bs[bn_f][bj + 0] = p;
                p.x = pack_bf2(v[4], v[5]); p.y = pack_bf2(v[6], v[7]);
                *(u32x2*)&Bs[bn_f][bj + 2] = p;
            }
        } else {
            const f32x4* bp = (const f32x4*)(Bp + (long)(nbase + bn_t) * ldb + k0 + bc_t);
            f32x4 f0 = bp[0], f1 = bp[1];
            const int c0 = bc_t >> 1;
            u32x2 p;
            p.x = pack_bf2(f0.x, f0.y); p.y = pack_bf2(f0.z, f0.w);
            *(u32x2*)&Bs[bn_t][c0 + 0] = p;
            p.x = pack_bf2(f1.x, f1.y); p.y = pack_bf2(f1.z, f1.w);
            *(u32x2*)&Bs[bn_t][c0 + 2] = p;
        }
        __syncthreads();

        // prefetch next A tile (global_prefetch_b8)
        if (k0 + 32 < K)
            __builtin_prefetch(A + (long)(mbase + ar) * lda + k0 + 32 + ac, 0, 1);

        // ---- fragments (4x ds_load_b64 each), reused 2x -> 4 WMMAs ----
        u32x8 afr[2], bfr[2];
#pragma unroll
        for (int mi = 0; mi < 2; ++mi) {
            const u32* arp = &As[32 * wm + 16 * mi + l][4 * hl];
            u32x2 a0 = *(const u32x2*)(arp + 0);   // K = 8*hl + 0..3
            u32x2 a1 = *(const u32x2*)(arp + 2);   // K = 8*hl + 4..7
            u32x2 a2 = *(const u32x2*)(arp + 8);   // K = 16 + 8*hl + 0..3
            u32x2 a3 = *(const u32x2*)(arp + 10);  // K = 16 + 8*hl + 4..7
            afr[mi] = u32x8{ a0.x, a0.y, a1.x, a1.y, a2.x, a2.y, a3.x, a3.y };
        }
#pragma unroll
        for (int ni = 0; ni < 2; ++ni) {
            const u32* brp = &Bs[32 * wn + 16 * ni + l][4 * hl];
            u32x2 b0 = *(const u32x2*)(brp + 0);   // K = 8*hl + 0..3
            u32x2 b1 = *(const u32x2*)(brp + 2);
            u32x2 b2 = *(const u32x2*)(brp + 8);
            u32x2 b3 = *(const u32x2*)(brp + 10);
            bfr[ni] = u32x8{ b0.x, b0.y, b1.x, b1.y, b2.x, b2.y, b3.x, b3.y };
        }
#pragma unroll
        for (int mi = 0; mi < 2; ++mi)
#pragma unroll
            for (int ni = 0; ni < 2; ++ni)
                acc[mi][ni] = __builtin_amdgcn_wmma_f32_16x16x32_bf16(
                                  false, __builtin_bit_cast(v16bf, afr[mi]),
                                  false, __builtin_bit_cast(v16bf, bfr[ni]),
                                  (short)0, acc[mi][ni], false, false);
        __syncthreads();
    }

    // ---- epilogue (C/D layout: rows r / r+8 per lane half) ----
#pragma unroll
    for (int mi = 0; mi < 2; ++mi) {
#pragma unroll
        for (int ni = 0; ni < 2; ++ni) {
            const int col  = nbase + 32 * wn + 16 * ni + l;
            const int row0 = mbase + 32 * wm + 16 * mi + 8 * hl;
            if (fullN || col < N) {
                float bv = bias ? bias[col] : 0.0f;
#pragma unroll
                for (int r = 0; r < 8; ++r) {
                    float v = acc[mi][ni][r] * alpha + bv;
                    if (ACT == 1) v = 0.5f * v * (1.0f + erff(v * 0.70710678118654752440f));
                    C[(long)(row0 + r) * ldc + col] = v;
                }
            }
        }
    }
}

// ---------------------------------------------------------------------------
// Embedding + positional encoding:  x = 2*emb[id] + pe(t)  ; pad mask
// ---------------------------------------------------------------------------
__global__ __launch_bounds__(256)
void embed_pe_kernel(const int* __restrict__ t1, const int* __restrict__ t2,
                     const float* __restrict__ emb, float* __restrict__ x,
                     int* __restrict__ pad)
{
    const int bt = blockIdx.x;           // 0..B*T-1
    const int b  = bt / T_, t = bt % T_;
    const int d  = threadIdx.x;
    const int half = L_ + 1;

    int id;
    if (t == 0 || t == half)      id = CLS_;
    else if (t < half)            id = t1[b * L_ + (t - 1)];
    else                          id = t2[b * L_ + (t - half - 1)];

    const int isPad = (id == 0);
    const int eid   = (id == MASK_ || id == 0) ? MASK_ : id;

    const float e = emb[(long)eid * D_ + d];
    const int   i2 = d >> 1;
    const float freq = __expf(-(float)(2 * i2) * (9.21034037197618f / (float)D_));
    const float ang  = (float)t * freq;
    const float pe   = (d & 1) ? cosf(ang) : sinf(ang);

    x[(long)bt * D_ + d] = 2.0f * e + pe;
    if (d == 0) pad[bt] = isPad;
}

// ---------------------------------------------------------------------------
// Masked scale + softmax in place over last dim.
// grid = (N rows, B*H) ; maskMode 0: pad[j]!=0 masked ; 1: ids[j]==0 masked
// ---------------------------------------------------------------------------
__global__ __launch_bounds__(256)
void softmax_kernel(float* __restrict__ sim, const int* __restrict__ mids,
                    int Mm, float scale, int maskMode)
{
    __shared__ float sh[8];
    const int z = blockIdx.y;            // b*H + h
    const int b = z / H_;
    const int i = blockIdx.x;
    float* row = sim + ((long)z * gridDim.x + i) * Mm;
    const int* mr = mids + (long)b * Mm;
    const int tid = threadIdx.x;

    float vals[8];
    int   cnt = 0;
    float mx = -3.402823466e38f;
    for (int j = tid; j < Mm; j += 256) {
        bool masked = maskMode ? (mr[j] == 0) : (mr[j] != 0);
        float v = masked ? -3.402823466e38f : row[j] * scale;
        vals[cnt++] = v;
        mx = fmaxf(mx, v);
    }
    mx = block_max(mx, sh);

    float sum = 0.0f;
    for (int i2 = 0; i2 < cnt; ++i2) {
        float e = __expf(vals[i2] - mx);
        vals[i2] = e;
        sum += e;
    }
    sum = block_sum(sum, sh);
    const float inv = 1.0f / sum;

    cnt = 0;
    for (int j = tid; j < Mm; j += 256) row[j] = vals[cnt++] * inv;
}

// ---------------------------------------------------------------------------
// out = LayerNorm(x + a) * g + be  ; one block (256 == D) per row
// ---------------------------------------------------------------------------
__global__ __launch_bounds__(256)
void add_ln_kernel(const float* __restrict__ x, const float* __restrict__ a,
                   const float* __restrict__ g, const float* __restrict__ be,
                   float* __restrict__ out)
{
    __shared__ float sh[8];
    const long row = blockIdx.x;
    const int  d   = threadIdx.x;
    float v = x[row * D_ + d] + a[row * D_ + d];
    const float mean = block_sum(v, sh) * (1.0f / (float)D_);
    const float c    = v - mean;
    const float var  = block_sum(c * c, sh) * (1.0f / (float)D_);
    out[row * D_ + d] = c * rsqrtf(var + 1e-5f) * g[d] + be[d];
}

// ---------------------------------------------------------------------------
// Pack outputs: logits[:,1:,:] then x[:,0,:]
// ---------------------------------------------------------------------------
__global__ __launch_bounds__(256)
void extract_kernel(const float* __restrict__ logits, const float* __restrict__ x3,
                    float* __restrict__ out)
{
    const long nlog = (long)B_ * (T_ - 1) * V_;
    const long idx  = (long)blockIdx.x * 256 + threadIdx.x;
    if (idx < nlog) {
        const long v = idx % V_;
        const long r = idx / V_;
        const long t = r % (T_ - 1) + 1;
        const long b = r / (T_ - 1);
        out[idx] = logits[((long)b * T_ + t) * V_ + v];
    } else if (idx < nlog + (long)B_ * D_) {
        const long k = idx - nlog;
        const long b = k / D_, d = k % D_;
        out[idx] = x3[(long)b * T_ * D_ + d];
    }
}

// ---------------------------------------------------------------------------
// Host orchestration
// ---------------------------------------------------------------------------
extern "C" void kernel_launch(void* const* d_in, const int* in_sizes, int n_in,
                              void* d_out, int out_size, void* d_ws, size_t ws_size,
                              hipStream_t stream)
{
    (void)in_sizes; (void)n_in; (void)out_size; (void)ws_size;

    const int*   src  = (const int*)  d_in[0];
    const int*   t1   = (const int*)  d_in[1];
    const int*   t2   = (const int*)  d_in[2];
    // d_in[3] original_lens: unused by the reference
    const float* enc  = (const float*)d_in[4];
    const float* emb  = (const float*)d_in[5];
    const float* Wq_s = (const float*)d_in[6];
    const float* Wk_s = (const float*)d_in[7];
    const float* Wv_s = (const float*)d_in[8];
    const float* Wo_s = (const float*)d_in[9];
    const float* bo_s = (const float*)d_in[10];
    const float* Wq_c = (const float*)d_in[11];
    const float* Wk_c = (const float*)d_in[12];
    const float* Wv_c = (const float*)d_in[13];
    const float* Wo_c = (const float*)d_in[14];
    const float* bo_c = (const float*)d_in[15];
    const float* W1   = (const float*)d_in[16];
    const float* b1   = (const float*)d_in[17];
    const float* W2   = (const float*)d_in[18];
    const float* b2   = (const float*)d_in[19];
    const float* g1   = (const float*)d_in[20];
    const float* g2   = (const float*)d_in[21];
    const float* g3   = (const float*)d_in[22];
    const float* be1  = (const float*)d_in[23];
    const float* be2  = (const float*)d_in[24];
    const float* be3  = (const float*)d_in[25];
    const float* Wfc  = (const float*)d_in[26];
    const float* bfc  = (const float*)d_in[27];
    float* out = (float*)d_out;

    float* W  = (float*)d_ws;
    float* X   = W;                 // [BT, D]  (also final X3)
    float* X1  = W + 1 * BTD_;
    float* X2  = W + 2 * BTD_;
    float* Y   = W + 3 * BTD_;      // proj / ffn out
    float* Y2  = W + 4 * BTD_;      // ffn hidden
    int*   PAD = (int*)(W + 5 * BTD_);  // [B*T]
    float* Q   = W + 6 * BTD_;      // [BT, INNER]; reused as attn-out and logits
    float* KB  = Q  + TI_;          // [BT, INNER] / [B*S, INNER]
    float* VB  = KB + TI_;
    float* SIM = VB + TI_;          // [B,H,T,T] (cross fits too)

    const float scale = 0.022097086912079608f;  // DH^-0.5
    const dim3 blk(256);

    // --- embedding + positional encoding ------------------------------------
    embed_pe_kernel<<<BT_, blk, 0, stream>>>(t1, t2, emb, X, PAD);

    // ======================= self-attention =================================
    gemm_wmma_kernel<false,0><<<dim3(INNER_/64, BT_/128, 1), blk, 0, stream>>>(
        X, Wq_s, nullptr, Q,  BT_, INNER_, D_, D_, INNER_, INNER_,
        1, 0,0, 0,0, 0,0, 1.0f);
    gemm_wmma_kernel<false,0><<<dim3(INNER_/64, BT_/128, 1), blk, 0, stream>>>(
        X, Wk_s, nullptr, KB, BT_, INNER_, D_, D_, INNER_, INNER_,
        1, 0,0, 0,0, 0,0, 1.0f);
    gemm_wmma_kernel<false,0><<<dim3(INNER_/64, BT_/128, 1), blk, 0, stream>>>(
        X, Wv_s, nullptr, VB, BT_, INNER_, D_, D_, INNER_, INNER_,
        1, 0,0, 0,0, 0,0, 1.0f);
    // sim[b,h] = Q[b,:,h,:] @ K[b,:,h,:]^T   (M=T, N=T, K=DH)
    gemm_wmma_kernel<true,0><<<dim3(T_/64, T_/128, B_*H_), blk, 0, stream>>>(
        Q, KB, nullptr, SIM, T_, T_, DH_, INNER_, INNER_, T_,
        H_, (long)T_*INNER_, DH_, (long)T_*INNER_, DH_,
        (long)H_*T_*T_, (long)T_*T_, 1.0f);
    softmax_kernel<<<dim3(T_, B_*H_), blk, 0, stream>>>(SIM, PAD, T_, scale, 0);
    // attn @ V -> Q region (Q dead)   (M=T, N=DH, K=T)
    gemm_wmma_kernel<false,0><<<dim3(DH_/64, T_/128, B_*H_), blk, 0, stream>>>(
        SIM, VB, nullptr, Q, T_, DH_, T_, T_, INNER_, INNER_,
        H_, (long)H_*T_*T_, (long)T_*T_, (long)T_*INNER_, DH_,
        (long)T_*INNER_, DH_, 1.0f);
    gemm_wmma_kernel<false,0><<<dim3(D_/64, BT_/128, 1), blk, 0, stream>>>(
        Q, Wo_s, bo_s, Y, BT_, D_, INNER_, INNER_, D_, D_,
        1, 0,0, 0,0, 0,0, 1.0f);
    add_ln_kernel<<<BT_, blk, 0, stream>>>(X, Y, g1, be1, X1);

    // ======================= cross-attention ================================
    gemm_wmma_kernel<false,0><<<dim3(INNER_/64, BT_/128, 1), blk, 0, stream>>>(
        X1, Wq_c, nullptr, Q, BT_, INNER_, D_, D_, INNER_, INNER_,
        1, 0,0, 0,0, 0,0, 1.0f);
    gemm_wmma_kernel<false,0><<<dim3(INNER_/64, (B_*S_)/128, 1), blk, 0, stream>>>(
        enc, Wk_c, nullptr, KB, B_*S_, INNER_, D_, D_, INNER_, INNER_,
        1, 0,0, 0,0, 0,0, 1.0f);
    gemm_wmma_kernel<false,0><<<dim3(INNER_/64, (B_*S_)/128, 1), blk, 0, stream>>>(
        enc, Wv_c, nullptr, VB, B_*S_, INNER_, D_, D_, INNER_, INNER_,
        1, 0,0, 0,0, 0,0, 1.0f);
    // sim[b,h] = Q @ Kc^T  (M=T, N=S, K=DH)
    gemm_wmma_kernel<true,0><<<dim3(S_/64, T_/128, B_*H_), blk, 0, stream>>>(
        Q, KB, nullptr, SIM, T_, S_, DH_, INNER_, INNER_, S_,
        H_, (long)T_*INNER_, DH_, (long)S_*INNER_, DH_,
        (long)H_*T_*S_, (long)T_*S_, 1.0f);
    softmax_kernel<<<dim3(T_, B_*H_), blk, 0, stream>>>(SIM, src, S_, scale, 1);
    gemm_wmma_kernel<false,0><<<dim3(DH_/64, T_/128, B_*H_), blk, 0, stream>>>(
        SIM, VB, nullptr, Q, T_, DH_, S_, S_, INNER_, INNER_,
        H_, (long)H_*T_*S_, (long)T_*S_, (long)S_*INNER_, DH_,
        (long)T_*INNER_, DH_, 1.0f);
    gemm_wmma_kernel<false,0><<<dim3(D_/64, BT_/128, 1), blk, 0, stream>>>(
        Q, Wo_c, bo_c, Y, BT_, D_, INNER_, INNER_, D_, D_,
        1, 0,0, 0,0, 0,0, 1.0f);
    add_ln_kernel<<<BT_, blk, 0, stream>>>(X1, Y, g2, be2, X2);

    // ======================= FFN (D -> D, exact GELU) =======================
    gemm_wmma_kernel<false,1><<<dim3(D_/64, BT_/128, 1), blk, 0, stream>>>(
        X2, W1, b1, Y2, BT_, D_, D_, D_, D_, D_, 1, 0,0, 0,0, 0,0, 1.0f);
    gemm_wmma_kernel<false,0><<<dim3(D_/64, BT_/128, 1), blk, 0, stream>>>(
        Y2, W2, b2, Y, BT_, D_, D_, D_, D_, D_, 1, 0,0, 0,0, 0,0, 1.0f);
    add_ln_kernel<<<BT_, blk, 0, stream>>>(X2, Y, g3, be3, X);   // X := X3

    // ======================= logits + output ===============================
    gemm_wmma_kernel<false,0><<<dim3((V_ + 63)/64, BT_/128, 1), blk, 0, stream>>>(
        X, Wfc, bfc, Q /* logits [BT,V] */, BT_, V_, D_, D_, V_, V_,
        1, 0,0, 0,0, 0,0, 1.0f);

    const long ntot = (long)B_ * (T_ - 1) * V_ + (long)B_ * D_;
    extract_kernel<<<(unsigned)((ntot + 255) / 256), blk, 0, stream>>>(Q, X, out);
}
